// TransformerLayer_11003706212889
// MI455X (gfx1250) — matrix operand order repaired and checked by reading, not compiled, not run on hardware
//
#include <hip/hip_runtime.h>
#include <hip/hip_bf16.h>
#include <cstdint>
#include <cstddef>

// B=4, S=1024, D=512, H=8, F=2048
#define BB 4
#define SS 1024
#define DD 512
#define HH 8
#define FF 2048

typedef __attribute__((ext_vector_type(16))) __bf16 v16bf;
typedef __attribute__((ext_vector_type(8)))  float  v8f;

// WMMA 16x16x32 bf16 fragment slot maps (this file's contiguous k order, k = 16*half + slot):
//  A (16x32, MxK): element (m,kk): half=kk>>4,     slot=kk&15,               lane=m+16*half
//  B (32x16, KxN): element (kk,n): half=kk>>4,     slot=kk&15,               lane=n+16*half
// => lane's 16-bf16 fragment is contiguous in a permuted buffer [tile][lane(32)][slot(16)].

__device__ __forceinline__ v8f tl_wmma(v16bf a, v16bf b, v8f c) {
  return __builtin_amdgcn_wmma_f32_16x16x32_bf16(false, a, false, b, (short)0, c, false, false);
}

// ---------------- fp32 -> bf16 conversion ----------------
__global__ void tl_cvt_bf16(const float* __restrict__ in, __bf16* __restrict__ out, int n) {
  int i = blockIdx.x * blockDim.x + threadIdx.x;
  if (i < n) out[i] = (__bf16)in[i];
}

// ---------------- weight permute: W[K][N] f32 -> B-fragment layout bf16 ----------------
// out[h][nt][kc][lane][slot], batch h = idx >> (nShift+kShift). Dims are powers of two.
__global__ void tl_permW(const float* __restrict__ W, __bf16* __restrict__ out,
                         int nShift, int kShift, int total) {
  int idx = blockIdx.x * blockDim.x + threadIdx.x;
  if (idx >= total) return;
  const int N = 1 << nShift, K = 1 << kShift;
  const int n = idx & (N - 1);
  const int k = (idx >> nShift) & (K - 1);
  const int h = idx >> (nShift + kShift);
  const int kk = k & 31, half = kk >> 4, slot = kk & 15;
  const int lane = (n & 15) + (half << 4);
  const int nt = n >> 4, kc = k >> 5, KC = K >> 5;
  const size_t o = ((size_t)h << (nShift + kShift)) +
                   ((((size_t)nt * KC + kc) << 9) + (lane << 4) + slot);
  out[o] = (__bf16)W[idx];
}

// ---------------- K permute for attention scores ----------------
// in: Kb [H][B*S][D] bf16 row-major.  out per (h,b): [kt(64)][c(16)][lane][slot]
// (B operand of Q@K^T: fragment k-dim = e-chunk of 32, n = key within 16-tile)
__global__ void tl_permK(const __bf16* __restrict__ in, __bf16* __restrict__ out, int total) {
  int idx = blockIdx.x * blockDim.x + threadIdx.x;
  if (idx >= total) return;
  const int d = idx & 511;
  const int r = idx >> 9;              // h*4096 + b*1024 + s
  const int s = r & 1023;
  const int hb = r >> 10;              // h*4 + b
  const int kt = s >> 4, kq = s & 15;
  const int c = d >> 5, kk = d & 31;
  const int half = kk >> 4, slot = kk & 15;
  const int lane = kq + (half << 4);
  const size_t o = ((size_t)hb << 19) + (((((size_t)kt << 4) + c) << 9) + (lane << 4) + slot);
  out[o] = in[idx];
}

// ---------------- V permute for attention PV ----------------
// out per (h,b): [kc(32)][ct(32)][lane][slot]
// (B operand of P@V: fragment k-dim = key within 32-chunk, n = e-col within 16-tile)
__global__ void tl_permV(const __bf16* __restrict__ in, __bf16* __restrict__ out, int total) {
  int idx = blockIdx.x * blockDim.x + threadIdx.x;
  if (idx >= total) return;
  const int d = idx & 511;
  const int r = idx >> 9;
  const int s = r & 1023;
  const int hb = r >> 10;
  const int kc = s >> 5, kk = s & 31;
  const int half = kk >> 4, slot = kk & 15;
  const int ct = d >> 4;
  const int lane = (d & 15) + (half << 4);
  const size_t o = ((size_t)hb << 19) + (((((size_t)kc << 5) + ct) << 9) + (lane << 4) + slot);
  out[o] = in[idx];
}

// ---------------- batched GEMM: C[z] = act(A[z] @ B[z] + bias[z]) ----------------
// 256 threads / 8 waves (2M x 4N); block tile 64x128; wave = 2x2 WMMA tiles.
// A staged double-buffered in permuted LDS (k-step 64); B fragments loaded
// directly from pre-permuted global weights (coalesced b128 per lane).
__global__ __launch_bounds__(256) void tl_gemm_bf16(
    const __bf16* __restrict__ A, unsigned long long strideA, int lda,
    const __bf16* __restrict__ Bp, unsigned long long strideB,  // permuted [nt][kc][32][16]
    const float* __restrict__ bias, unsigned long long strideBias,
    __bf16* __restrict__ Cb, float* __restrict__ Cf, unsigned long long strideC, int ldc,
    int M, int N, int K, int relu) {
  const int tid = threadIdx.x;
  const int w = tid >> 5, lane = tid & 31;
  const int wm = w >> 2, wn = w & 3;
  const int bm = blockIdx.y * 64, bn = blockIdx.x * 128;
  const size_t z = blockIdx.z;
  A += z * (size_t)strideA;
  Bp += z * (size_t)strideB;
  if (bias) bias += z * (size_t)strideBias;
  if (Cb) Cb += z * (size_t)strideC;
  if (Cf) Cf += z * (size_t)strideC;
  const int KC = K >> 5;

  // Ap[buf][kcL(2)][m-tile(4)][lane(32)][slot(16)]
  __shared__ alignas(32) __bf16 Ap[2][2 * 4 * 32 * 16];

  const v8f vzero = {0.f, 0.f, 0.f, 0.f, 0.f, 0.f, 0.f, 0.f};
  v8f acc[2][2];
  acc[0][0] = vzero; acc[0][1] = vzero; acc[1][0] = vzero; acc[1][1] = vzero;

  auto stageA = [&](int buf, int k0) {
    // A block 64(M) x 64(K): 1024 quads, 4 per thread
#pragma unroll
    for (int it = 0; it < 4; ++it) {
      const int q = tid + (it << 8);
      const int m = q >> 4;
      const int e = (q & 15) << 2;
      const uint2 d = *(const uint2*)(A + (size_t)(bm + m) * lda + k0 + e);
      const int kcL = e >> 5, kk = e & 31;
      const int half = kk >> 4;
      const int slot = kk & 15;
      const int mt = m >> 4, mm = m & 15;
      *(uint2*)&Ap[buf][(((((kcL << 2) + mt) << 5) + mm + (half << 4)) << 4) + slot] = d;
    }
  };

  const int nt0 = (bn >> 4) + (wn << 1);
  stageA(0, 0);
  int p = 0;
  for (int k0 = 0; k0 < K; k0 += 64) {
    __syncthreads();
    if (k0 + 64 < K) stageA(p ^ 1, k0 + 64);
    if (k0 + 128 < K)
      __builtin_prefetch(A + (size_t)(bm + (tid >> 2)) * lda + k0 + 128, 0, 1);
#pragma unroll
    for (int kcL = 0; kcL < 2; ++kcL) {
      const int kc = (k0 >> 5) + kcL;
      const __bf16* ap = &Ap[p][(((kcL << 2) << 5) << 4)];
      const v16bf a0 = *(const v16bf*)(ap + ((((0 + (wm << 1)) << 5) + lane) << 4));
      const v16bf a1 = *(const v16bf*)(ap + ((((1 + (wm << 1)) << 5) + lane) << 4));
      const v16bf b0 = *(const v16bf*)(Bp + ((((size_t)nt0 * KC + kc) << 9) + (lane << 4)));
      const v16bf b1 = *(const v16bf*)(Bp + ((((size_t)(nt0 + 1) * KC + kc) << 9) + (lane << 4)));
      acc[0][0] = tl_wmma(a0, b0, acc[0][0]);
      acc[0][1] = tl_wmma(a0, b1, acc[0][1]);
      acc[1][0] = tl_wmma(a1, b0, acc[1][0]);
      acc[1][1] = tl_wmma(a1, b1, acc[1][1]);
    }
    p ^= 1;
  }

  const int n = lane & 15, half = (lane >> 4) & 1;
#pragma unroll
  for (int ti = 0; ti < 2; ++ti) {
#pragma unroll
    for (int tj = 0; tj < 2; ++tj) {
      const int col = bn + ((wn << 1) + tj) * 16 + n;
      const float bv = bias ? bias[col] : 0.f;
#pragma unroll
      for (int r = 0; r < 8; ++r) {
        const int row = bm + ((wm << 1) + ti) * 16 + r + 8 * half;
        float v = acc[ti][tj][r] + bv;
        if (relu) v = fmaxf(v, 0.f);
        if (Cb) Cb[(size_t)row * ldc + col] = (__bf16)v;
        if (Cf) Cf[(size_t)row * ldc + col] = v;
      }
    }
  }
}

// ---------------- Flash attention ----------------
// grid: (S/16, H, B). 4 waves; scores: wave w owns one 16-key tile of the 64-key
// block; PV: wave w owns 128 E-columns. K/V fragments direct from permuted global.
__global__ __launch_bounds__(128) void tl_attn(
    const __bf16* __restrict__ Qb, const __bf16* __restrict__ Kp,
    const __bf16* __restrict__ Vp, __bf16* __restrict__ Cat) {
  const int q0 = blockIdx.x * 16;
  const int h = blockIdx.y;
  const int b = blockIdx.z;
  const int tid = threadIdx.x;
  const int w = tid >> 5;
  const int lane = tid & 31;

  const __bf16* Qh = Qb + (((size_t)h * (BB * SS) + (size_t)b * SS) + q0) * DD;
  const __bf16* Kh = Kp + (((size_t)h << 2) + b) * (1u << 19);
  const __bf16* Vh = Vp + (((size_t)h << 2) + b) * (1u << 19);

  // Qp: [e-chunk(16)][lane(32)][slot(16)] (A layout) 16 KB; Pp: 2 KB
  __shared__ alignas(32) __bf16 Qs[16 * 32 * 16];
  __shared__ float Ss[16][64];
  __shared__ alignas(32) __bf16 Pp[2 * 32 * 16];
  __shared__ float mrow[16], lrow[16], arow[16];

  // stage Q permuted: 2048 quads, 16 per thread
  for (int q = tid; q < 2048; q += 128) {
    const int r = q >> 7;
    const int e = (q & 127) << 2;
    const uint2 d = *(const uint2*)(Qh + (size_t)r * DD + e);
    const int c = e >> 5, kk = e & 31;
    const int half = kk >> 4;
    const int slot = kk & 15;
    *(uint2*)&Qs[(((c << 5) + r + (half << 4)) << 4) + slot] = d;
  }
  if (tid < 16) { mrow[tid] = -3.0e38f; lrow[tid] = 0.f; }
  __syncthreads();

  const v8f vzero = {0.f, 0.f, 0.f, 0.f, 0.f, 0.f, 0.f, 0.f};
  v8f acc[8];
#pragma unroll
  for (int j = 0; j < 8; ++j) acc[j] = vzero;

  const float scale = 0.044194173824159216f;  // 1/sqrt(512)

  for (int t0 = 0; t0 < SS; t0 += 64) {
    // --- scores: sacc(16q x 16k) = Q(16x512) @ K^T(512x16), kt = t0/16 + w ---
    v8f sacc = vzero;
    const int kt = (t0 >> 4) + w;
#pragma unroll
    for (int c = 0; c < 16; ++c) {
      const v16bf a = *(const v16bf*)&Qs[(((c << 5) + lane) << 4)];
      const v16bf kb = *(const v16bf*)(Kh + (((((size_t)kt << 4) + c) << 9) + (lane << 4)));
      sacc = tl_wmma(a, kb, sacc);
    }
    if (t0 + 64 < SS)
      __builtin_prefetch(Kh + ((((size_t)(kt + 4) << 4) + (tid & 15)) << 9), 0, 1);
    {
      const int n = lane & 15, half = (lane >> 4) & 1;
#pragma unroll
      for (int r = 0; r < 8; ++r)
        Ss[r + 8 * half][w * 16 + n] = sacc[r] * scale;
    }
    __syncthreads();
    // --- online softmax (one thread per query row), P written in A layout ---
    if (tid < 16) {
      float mb = -3.0e38f;
      for (int jj = 0; jj < 64; ++jj) mb = fmaxf(mb, Ss[tid][jj]);
      const float mnew = fmaxf(mrow[tid], mb);
      const float al = __expf(mrow[tid] - mnew);
      float sum = 0.f;
      for (int jj = 0; jj < 64; ++jj) {
        const float pv = __expf(Ss[tid][jj] - mnew);
        const int c = jj >> 5, kk = jj & 31;
        const int half = kk >> 4;
        const int slot = kk & 15;
        Pp[(((c << 5) + tid + (half << 4)) << 4) + slot] = (__bf16)pv;
        sum += pv;
      }
      lrow[tid] = lrow[tid] * al + sum;
      mrow[tid] = mnew;
      arow[tid] = al;
    }
    __syncthreads();
    // --- rescale running O by alpha ---
    {
      const int half = (lane >> 4) & 1;
#pragma unroll
      for (int j = 0; j < 8; ++j)
#pragma unroll
        for (int r = 0; r < 8; ++r)
          acc[j][r] *= arow[r + 8 * half];
    }
    // --- PV: O[16 x 128] += P(16x64) @ V(64x128) for this wave's E slice ---
#pragma unroll
    for (int kkc = 0; kkc < 2; ++kkc) {
      const v16bf a = *(const v16bf*)&Pp[(((kkc << 5) + lane) << 4)];
      const int kc = (t0 >> 5) + kkc;
#pragma unroll
      for (int j = 0; j < 8; ++j) {
        const int ct = (w << 3) + j;
        const v16bf vb = *(const v16bf*)(Vh + (((((size_t)kc << 5) + ct) << 9) + (lane << 4)));
        acc[j] = tl_wmma(a, vb, acc[j]);
      }
    }
  }

  // --- epilogue: divide by l, scatter into concat [B*S, H*D] ---
  {
    const int n = lane & 15, half = (lane >> 4) & 1;
#pragma unroll
    for (int j = 0; j < 8; ++j) {
#pragma unroll
      for (int r = 0; r < 8; ++r) {
        const int m = r + 8 * half;
        const float o = acc[j][r] / lrow[m];
        const size_t row = (size_t)b * SS + q0 + m;
        const size_t col = (size_t)h * DD + w * 128 + j * 16 + n;
        Cat[row * (HH * DD) + col] = (__bf16)o;
      }
    }
  }
}

// ---------------- LayerNorm( a + bres ) * g + beta ----------------
__global__ __launch_bounds__(256) void tl_ln(
    const float* __restrict__ A, const float* __restrict__ Bres,
    const float* __restrict__ g, const float* __restrict__ beta,
    float* __restrict__ outF, __bf16* __restrict__ outB) {
  const int row = blockIdx.x;
  const int tid = threadIdx.x;
  __shared__ float s1[256], s2[256];
  const float* a = A + (size_t)row * DD;
  const float* b = Bres + (size_t)row * DD;
  const float x0 = a[tid] + b[tid];
  const float x1 = a[tid + 256] + b[tid + 256];
  s1[tid] = x0 + x1;
  s2[tid] = x0 * x0 + x1 * x1;
  __syncthreads();
  for (int off = 128; off > 0; off >>= 1) {
    if (tid < off) { s1[tid] += s1[tid + off]; s2[tid] += s2[tid + off]; }
    __syncthreads();
  }
  const float mean = s1[0] * (1.f / DD);
  const float var = s2[0] * (1.f / DD) - mean * mean;
  const float rstd = rsqrtf(var + 1e-6f);
  const float y0 = (x0 - mean) * rstd * g[tid] + beta[tid];
  const float y1 = (x1 - mean) * rstd * g[tid + 256] + beta[tid + 256];
  if (outF) {
    outF[(size_t)row * DD + tid] = y0;
    outF[(size_t)row * DD + tid + 256] = y1;
  }
  if (outB) {
    outB[(size_t)row * DD + tid] = (__bf16)y0;
    outB[(size_t)row * DD + tid + 256] = (__bf16)y1;
  }
}

// ---------------- host-side orchestration ----------------
extern "C" void kernel_launch(void* const* d_in, const int* in_sizes, int n_in,
                              void* d_out, int out_size, void* d_ws, size_t ws_size,
                              hipStream_t stream) {
  (void)in_sizes; (void)n_in; (void)out_size; (void)ws_size;
  const float* x  = (const float*)d_in[0];
  const float* Wq = (const float*)d_in[1];
  const float* bq = (const float*)d_in[2];
  const float* Wk = (const float*)d_in[3];
  const float* bk = (const float*)d_in[4];
  const float* Wv = (const float*)d_in[5];
  const float* bv = (const float*)d_in[6];
  const float* Wo = (const float*)d_in[7];
  const float* bo = (const float*)d_in[8];
  const float* g1 = (const float*)d_in[9];
  const float* be1 = (const float*)d_in[10];
  const float* W1 = (const float*)d_in[11];
  const float* b1 = (const float*)d_in[12];
  const float* W2 = (const float*)d_in[13];
  const float* b2 = (const float*)d_in[14];
  const float* g2 = (const float*)d_in[15];
  const float* be2 = (const float*)d_in[16];
  float* out = (float*)d_out;

  const size_t R = (size_t)BB * SS;  // 4096 rows
  char* ws = (char*)d_ws;
  size_t off = 0;
  auto take = [&](size_t bytes) -> char* {
    char* p = ws + off;
    off += (bytes + 255) & ~(size_t)255;
    return p;
  };
  __bf16* xb  = (__bf16*)take(R * DD * 2);
  __bf16* Wqp = (__bf16*)take((size_t)HH * DD * DD * 2);
  __bf16* Wkp = (__bf16*)take((size_t)HH * DD * DD * 2);
  __bf16* Wvp = (__bf16*)take((size_t)HH * DD * DD * 2);
  __bf16* Wop = (__bf16*)take((size_t)HH * DD * DD * 2);
  __bf16* W1p = (__bf16*)take((size_t)DD * FF * 2);
  __bf16* W2p = (__bf16*)take((size_t)FF * DD * 2);
  __bf16* Qb  = (__bf16*)take((size_t)HH * R * DD * 2);
  __bf16* Kb  = (__bf16*)take((size_t)HH * R * DD * 2);
  __bf16* Vb  = (__bf16*)take((size_t)HH * R * DD * 2);
  __bf16* Kpg = (__bf16*)take((size_t)HH * R * DD * 2);
  __bf16* Vpg = (__bf16*)take((size_t)HH * R * DD * 2);
  __bf16* Cat = (__bf16*)take(R * (size_t)(HH * DD) * 2);
  float*  mha = (float*)take(R * DD * 4);
  float*  yf  = (float*)take(R * DD * 4);
  __bf16* yb  = (__bf16*)take(R * DD * 2);
  __bf16* h1b = (__bf16*)take(R * (size_t)FF * 2);
  float*  ffo = (float*)take(R * DD * 4);

  // 1) convert x; permute+convert weights into B-fragment layout
  tl_cvt_bf16<<<(unsigned)((R * DD + 255) / 256), 256, 0, stream>>>(x, xb, (int)(R * DD));
  tl_permW<<<(HH * DD * DD + 255) / 256, 256, 0, stream>>>(Wq, Wqp, 9, 9, HH * DD * DD);
  tl_permW<<<(HH * DD * DD + 255) / 256, 256, 0, stream>>>(Wk, Wkp, 9, 9, HH * DD * DD);
  tl_permW<<<(HH * DD * DD + 255) / 256, 256, 0, stream>>>(Wv, Wvp, 9, 9, HH * DD * DD);
  tl_permW<<<(HH * DD * DD + 255) / 256, 256, 0, stream>>>(Wo, Wop, 9, 12, HH * DD * DD);
  tl_permW<<<(DD * FF + 255) / 256, 256, 0, stream>>>(W1, W1p, 11, 9, DD * FF);
  tl_permW<<<(FF * DD + 255) / 256, 256, 0, stream>>>(W2, W2p, 9, 11, FF * DD);

  // 2) QKV projections, batched over heads
  {
    dim3 grid(DD / 128, (unsigned)(R / 64), HH);
    tl_gemm_bf16<<<grid, 256, 0, stream>>>(
        xb, 0ull, DD, Wqp, (unsigned long long)DD * DD, bq, (unsigned long long)DD,
        Qb, nullptr, (unsigned long long)(R * DD), DD, (int)R, DD, DD, 0);
    tl_gemm_bf16<<<grid, 256, 0, stream>>>(
        xb, 0ull, DD, Wkp, (unsigned long long)DD * DD, bk, (unsigned long long)DD,
        Kb, nullptr, (unsigned long long)(R * DD), DD, (int)R, DD, DD, 0);
    tl_gemm_bf16<<<grid, 256, 0, stream>>>(
        xb, 0ull, DD, Wvp, (unsigned long long)DD * DD, bv, (unsigned long long)DD,
        Vb, nullptr, (unsigned long long)(R * DD), DD, (int)R, DD, DD, 0);
  }

  // 3) permute K, V into attention fragment layouts
  {
    const int total = HH * BB * SS * DD;
    tl_permK<<<total / 256, 256, 0, stream>>>(Kb, Kpg, total);
    tl_permV<<<total / 256, 256, 0, stream>>>(Vb, Vpg, total);
  }

  // 4) flash attention -> Cat [B*S, H*D] (bf16)
  {
    dim3 grid(SS / 16, HH, BB);
    tl_attn<<<grid, 128, 0, stream>>>(Qb, Kpg, Vpg, Cat);
  }

  // 5) output projection: mha = Cat @ Wo + bo (fp32 out)
  {
    dim3 grid(DD / 128, (unsigned)(R / 64), 1);
    tl_gemm_bf16<<<grid, 256, 0, stream>>>(
        Cat, 0ull, HH * DD, Wop, 0ull, bo, 0ull,
        nullptr, mha, 0ull, DD, (int)R, DD, HH * DD, 0);
  }

  // 6) y = LN(x + mha)
  tl_ln<<<(unsigned)R, 256, 0, stream>>>(x, mha, g1, be1, yf, yb);

  // 7) h1 = relu(y @ W1 + b1) (bf16 out)
  {
    dim3 grid(FF / 128, (unsigned)(R / 64), 1);
    tl_gemm_bf16<<<grid, 256, 0, stream>>>(
        yb, 0ull, DD, W1p, 0ull, b1, 0ull,
        h1b, nullptr, 0ull, FF, (int)R, FF, DD, 1);
  }

  // 8) ff = h1 @ W2 + b2 (fp32 out)
  {
    dim3 grid(DD / 128, (unsigned)(R / 64), 1);
    tl_gemm_bf16<<<grid, 256, 0, stream>>>(
        h1b, 0ull, FF, W2p, 0ull, b2, 0ull,
        nullptr, ffo, 0ull, DD, (int)R, DD, FF, 0);
  }

  // 9) out = LN(y + ff)
  tl_ln<<<(unsigned)R, 256, 0, stream>>>(yf, ffo, g2, be2, out, nullptr);
}
